// DynamicProxyNCA_41429254537747
// MI455X (gfx1250) — matrix operand order repaired and checked by reading, not compiled, not run on hardware
//
#include <hip/hip_runtime.h>
#include <math.h>

// Problem constants (from reference)
#define B_N   8192      // samples
#define Z_D   128       // feature dim
#define NCLS  62        // classes
#define NPRX  93        // proxies
#define PPAD  96        // proxies padded to 6 WMMA column tiles
#define NA    2730      // anchors = len(arange(0, 8189, 3))
#define APAD  2736      // anchors padded to 171*16
#define MT    (APAD/16) // 171 anchor row-tiles
#define EPSV  1e-6f
#define EPS2T (2.0f*EPSV)
#define ZEPS2 (128.0f*EPSV*EPSV)

typedef __attribute__((ext_vector_type(16))) _Float16 v16h;
typedef __attribute__((ext_vector_type(8)))  float    v8f;

// ---------------------------------------------------------------- helpers ---

__device__ __forceinline__ unsigned long long shfl_xor_u64(unsigned long long v, int mask) {
    int lo = (int)(unsigned)(v & 0xFFFFFFFFull);
    int hi = (int)(unsigned)(v >> 32);
    lo = __shfl_xor(lo, mask, 32);
    hi = __shfl_xor(hi, mask, 32);
    return ((unsigned long long)(unsigned)hi << 32) | (unsigned)lo;
}

// A-operand loader from pre-converted f16 rows (row length 128 halves, 16B aligned).
// Layout (matches B loader): lanes 0-15 hold K {kk..kk+15},
// lanes 16-31 hold K {kk+16..kk+31}. Both 8-half groups are 16B-aligned
// contiguous chunks -> two global_load_b128.
__device__ __forceinline__ v16h load_a16h(const _Float16* arow, int kk, int half) {
    union { uint4 u[2]; v16h v; } cvt;
    cvt.u[0] = *reinterpret_cast<const uint4*>(arow + kk + half * 16);
    cvt.u[1] = *reinterpret_cast<const uint4*>(arow + kk + 8 + half * 16);
    return cvt.v;
}

// B-operand loader: lane n (0-15) holds column n with K = kk..kk+15 (contiguous in
// the source row); lanes 16-31 hold K = kk+16..kk+31. 32B contiguous -> 2x b128.
__device__ __forceinline__ v16h load_b16h(const _Float16* brow, int kk, int half) {
    const uint4* p = reinterpret_cast<const uint4*>(brow + kk + half * 16);
    union { uint4 u[2]; v16h v; } cvt;
    cvt.u[0] = p[0];
    cvt.u[1] = p[1];
    return cvt.v;
}

// ---------------------------------------------------------------- kernels ---

__global__ void k_init(unsigned long long* pmin, unsigned long long* pmax, float* loss) {
    int i = blockIdx.x * blockDim.x + threadIdx.x;
    if (i < APAD) { pmin[i] = ~0ull; pmax[i] = 0ull; loss[i] = 0.0f; }
}

// Per-sample: sum(z_j), sum(z_j^2), f16 mirror of z, and
// y_rel[j] = first c with y_map[c]==y_idx[j].
__global__ void k_prep_z(const float* __restrict__ z, const int* __restrict__ y_idx,
                         const int* __restrict__ y_map,
                         float* zz, float* zsum, int* yrel, _Float16* zh) {
    __shared__ float s1[128], s2[128];
    int j = blockIdx.x, k = threadIdx.x;
    float v = z[(size_t)j * Z_D + k];
    zh[(size_t)j * Z_D + k] = (_Float16)v;
    s1[k] = v; s2[k] = v * v;
    __syncthreads();
    for (int s = 64; s > 0; s >>= 1) {
        if (k < s) { s1[k] += s1[k + s]; s2[k] += s2[k + s]; }
        __syncthreads();
    }
    if (k == 0) {
        zsum[j] = s1[0]; zz[j] = s2[0];
        int yi = y_idx[j], r = 0;
        for (int c = NCLS - 1; c >= 0; --c) if (y_map[c] == yi) r = c;  // first match
        yrel[j] = r;
    }
}

// Normalize proxies (clamped norm), store f32 + f16 normalized rows (padded to 96),
// plus per-proxy sum and squared-norm.
__global__ void k_prep_prx(const float* __restrict__ proxies,
                           float* prx, _Float16* prxh, float* psum, float* pnrm2) {
    __shared__ float s1[128], s2[128];
    int p = blockIdx.x, k = threadIdx.x;
    float v = (p < NPRX) ? proxies[(size_t)p * Z_D + k] : 0.0f;
    s2[k] = v * v;
    __syncthreads();
    for (int s = 64; s > 0; s >>= 1) { if (k < s) s2[k] += s2[k + s]; __syncthreads(); }
    float nrm = fmaxf(sqrtf(s2[0]), 1e-12f);
    __syncthreads();
    float nv = v / nrm;
    prx[(size_t)p * Z_D + k]  = nv;
    prxh[(size_t)p * Z_D + k] = (_Float16)nv;
    s1[k] = nv; s2[k] = nv * nv;
    __syncthreads();
    for (int s = 64; s > 0; s >>= 1) {
        if (k < s) { s1[k] += s1[k + s]; s2[k] += s2[k + s]; }
        __syncthreads();
    }
    if (k == 0) { psum[p] = s1[0]; pnrm2[p] = s2[0]; }
}

// Stage 1 WMMA: nearest proxy per anchor. One wave computes a 16-anchor x 96-proxy
// strip: per K-step the A fragment is loaded once and fed to 6 B tiles
// (24 back-to-back v_wmma). Packed-u64 argmin fused in the epilogue
// (ties -> smallest proxy index, matching jnp.argmin).
__global__ void k_nearest_proxy(const _Float16* __restrict__ zh, const _Float16* __restrict__ prxh,
                                const float* __restrict__ zz, const float* __restrict__ zsum,
                                const float* __restrict__ psum, const float* __restrict__ pnrm2,
                                unsigned long long* pmin) {
    int lane = threadIdx.x & 31;
    int half = lane >> 4;
    int m    = lane & 15;
    int aBase = blockIdx.x * 16;

    int aRow = aBase + m;
    const _Float16* arow = zh + (size_t)(3 * (aRow < NA ? aRow : NA - 1)) * Z_D;

    v8f c[6];
#pragma unroll
    for (int t = 0; t < 6; ++t) c[t] = (v8f){};

#pragma unroll
    for (int ks = 0; ks < 4; ++ks) {
        v16h a = load_a16h(arow, ks * 32, half);
#pragma unroll
        for (int t = 0; t < 6; ++t) {
            v16h b = load_b16h(prxh + (size_t)(t * 16 + m) * Z_D, ks * 32, half);
            c[t] = __builtin_amdgcn_wmma_f32_16x16x32_f16(false, a, false, b, (short)0, c[t],
                                                          false, false);
        }
    }

    // Hoist per-anchor-row terms (same aIdx for all 6 column tiles).
    float azz[8], azs[8];
    bool  aval[8];
#pragma unroll
    for (int r = 0; r < 8; ++r) {
        int aIdx = aBase + half * 8 + r;
        int j = 3 * (aIdx < NA ? aIdx : NA - 1);
        azz[r] = zz[j]; azs[r] = zsum[j]; aval[r] = (aIdx < NA);
    }

    unsigned long long best[8];
#pragma unroll
    for (int r = 0; r < 8; ++r) best[r] = ~0ull;
#pragma unroll
    for (int t = 0; t < 6; ++t) {
        int n = t * 16 + m;                    // proxy column for this lane
        float bb = pnrm2[n], bs = psum[n];
        bool nval = (n < NPRX);
#pragma unroll
        for (int r = 0; r < 8; ++r) {
            float d2 = azz[r] + bb - 2.0f * c[t][r] + EPS2T * (azs[r] - bs) + ZEPS2;
            float d  = sqrtf(fmaxf(d2, 0.0f));
            unsigned long long pk = (aval[r] && nval)
                ? ((((unsigned long long)__float_as_uint(d)) << 32) | (unsigned)n) : ~0ull;
            if (pk < best[r]) best[r] = pk;
        }
    }
#pragma unroll
    for (int mk = 1; mk < 16; mk <<= 1) {
#pragma unroll
        for (int r = 0; r < 8; ++r) {
            unsigned long long o = shfl_xor_u64(best[r], mk);
            if (o < best[r]) best[r] = o;
        }
    }
    if (m == 0) {
#pragma unroll
        for (int r = 0; r < 8; ++r) {
            int aIdx = aBase + half * 8 + r;
            if (aIdx < NA) atomicMin(&pmin[aIdx], best[r]);
        }
    }
}

// Gather per-anchor chosen-proxy metadata + anchor class.
__global__ void k_gather_anchor(const unsigned long long* __restrict__ pmin,
                                const float* __restrict__ psum, const float* __restrict__ pnrm2,
                                const int* __restrict__ yrel,
                                int* pidx, float* a_sum, float* a_nrm2, int* a_cls) {
    int a = blockIdx.x * blockDim.x + threadIdx.x;
    if (a >= APAD) return;
    if (a < NA) {
        int pi = (int)(unsigned)(pmin[a] & 0xFFFFFFFFull);
        pi = min(max(pi, 0), NPRX - 1);
        pidx[a] = pi; a_sum[a] = psum[pi]; a_nrm2[a] = pnrm2[pi]; a_cls[a] = yrel[3 * a];
    } else {
        pidx[a] = 0; a_sum[a] = 0.0f; a_nrm2[a] = 0.0f; a_cls[a] = -1;
    }
}

// Stage 2 WMMA (dominant GEMM): D_pz = pdist(prox_p, z) over [2736,8192].
// One wave computes a 16-anchor x 64-column strip: A fragment loaded once per
// K-step, 4 B tiles -> 16 v_wmma per wave with 4x A reuse. Masked
// (j >= 3a && same class) packed-u64 argmax fused into the epilogue
// (ties -> smallest j via complemented index, matching jnp.argmax).
__global__ void k_hard_pos(const _Float16* __restrict__ zh, const _Float16* __restrict__ prxh,
                           const float* __restrict__ zz, const float* __restrict__ zsum,
                           const int* __restrict__ yrel, const int* __restrict__ pidx,
                           const float* __restrict__ a_sum, const float* __restrict__ a_nrm2,
                           const int* __restrict__ a_cls, unsigned long long* pmax) {
    int lane = threadIdx.x & 31;
    int half = lane >> 4;
    int m    = lane & 15;
    int aBase = blockIdx.x * 16;
    int jBase = blockIdx.y * 64;

    // Overlap epilogue-metadata fetch with the GEMM.
    __builtin_prefetch(zz + jBase + m, 0, 0);
    __builtin_prefetch(zsum + jBase + m, 0, 0);
    __builtin_prefetch(yrel + jBase + m, 0, 0);

    const _Float16* arow = prxh + (size_t)pidx[aBase + m] * Z_D;   // chosen proxy row

    v8f c[4];
#pragma unroll
    for (int t = 0; t < 4; ++t) c[t] = (v8f){};

#pragma unroll
    for (int ks = 0; ks < 4; ++ks) {
        v16h a = load_a16h(arow, ks * 32, half);
#pragma unroll
        for (int t = 0; t < 4; ++t) {
            v16h b = load_b16h(zh + (size_t)(jBase + t * 16 + m) * Z_D, ks * 32, half);
            c[t] = __builtin_amdgcn_wmma_f32_16x16x32_f16(false, a, false, b, (short)0, c[t],
                                                          false, false);
        }
    }

    // Hoist per-anchor-row terms (same aIdx for all 4 column tiles).
    float an2[8], asm_[8];
    int   acl[8], a3[8];
    bool  aval[8];
#pragma unroll
    for (int r = 0; r < 8; ++r) {
        int aIdx = aBase + half * 8 + r;
        an2[r] = a_nrm2[aIdx]; asm_[r] = a_sum[aIdx]; acl[r] = a_cls[aIdx];
        a3[r] = 3 * aIdx; aval[r] = (aIdx < NA);
    }

    unsigned long long best[8];
#pragma unroll
    for (int r = 0; r < 8; ++r) best[r] = 0ull;
#pragma unroll
    for (int t = 0; t < 4; ++t) {
        int j = jBase + t * 16 + m;
        float bb = zz[j], bs = zsum[j];
        int   yj = yrel[j];
#pragma unroll
        for (int r = 0; r < 8; ++r) {
            float d2 = an2[r] + bb - 2.0f * c[t][r] + EPS2T * (asm_[r] - bs) + ZEPS2;
            float d  = sqrtf(fmaxf(d2, 0.0f));
            bool valid = aval[r] && (j >= a3[r]) && (yj == acl[r]);
            unsigned long long pk = valid
                ? ((((unsigned long long)__float_as_uint(d)) << 32) | (unsigned)(~(unsigned)j))
                : 0ull;
            if (pk > best[r]) best[r] = pk;
        }
    }
#pragma unroll
    for (int mk = 1; mk < 16; mk <<= 1) {
#pragma unroll
        for (int r = 0; r < 8; ++r) {
            unsigned long long o = shfl_xor_u64(best[r], mk);
            if (o > best[r]) best[r] = o;
        }
    }
    if (m == 0) {
#pragma unroll
        for (int r = 0; r < 8; ++r) {
            int aIdx = aBase + half * 8 + r;
            if (aIdx < NA) atomicMax(&pmax[aIdx], best[r]);
        }
    }
}

// Stage 3: per anchor, D_n = pdist(z[j_p], proxies); loss = D_p + logsumexp(-D_n).
__global__ void k_loss(const float* __restrict__ z, const float* __restrict__ prx,
                       const float* __restrict__ zz, const float* __restrict__ zsum,
                       const float* __restrict__ psum, const float* __restrict__ pnrm2,
                       const unsigned long long* __restrict__ pmax, float* loss) {
    __shared__ float zr[128];
    __shared__ float red[128];
    int a = blockIdx.x, t = threadIdx.x;
    unsigned long long pk = pmax[a];
    int jp = (int)(~(unsigned)(pk & 0xFFFFFFFFull));
    jp = min(max(jp, 0), B_N - 1);
    float Dp = __uint_as_float((unsigned)(pk >> 32));
    zr[t] = z[(size_t)jp * Z_D + t];
    __syncthreads();

    float nd = -INFINITY;
    if (t < NPRX) {
        const float* pr = prx + (size_t)t * Z_D;
        float dot = 0.0f;
#pragma unroll 8
        for (int k = 0; k < Z_D; ++k) dot = fmaf(zr[k], pr[k], dot);
        float d2 = zz[jp] + pnrm2[t] - 2.0f * dot + EPS2T * (zsum[jp] - psum[t]) + ZEPS2;
        nd = -sqrtf(fmaxf(d2, 0.0f));
    }
    red[t] = nd;
    __syncthreads();
    for (int s = 64; s > 0; s >>= 1) { if (t < s) red[t] = fmaxf(red[t], red[t + s]); __syncthreads(); }
    float mx = red[0];
    __syncthreads();
    red[t] = (t < NPRX) ? expf(nd - mx) : 0.0f;
    __syncthreads();
    for (int s = 64; s > 0; s >>= 1) { if (t < s) red[t] += red[t + s]; __syncthreads(); }
    if (t == 0) loss[a] = Dp + mx + logf(red[0]);
}

// Deterministic final mean (fixed-order strided sums + LDS tree).
__global__ void k_reduce(const float* __restrict__ loss, float* out) {
    __shared__ float red[256];
    int t = threadIdx.x;
    float s = 0.0f;
    for (int a = t; a < NA; a += 256) s += loss[a];
    red[t] = s;
    __syncthreads();
    for (int st = 128; st > 0; st >>= 1) { if (t < st) red[t] += red[t + st]; __syncthreads(); }
    if (t == 0) out[0] = red[0] / (float)NA;
}

// ----------------------------------------------------------------- launch ---

extern "C" void kernel_launch(void* const* d_in, const int* in_sizes, int n_in,
                              void* d_out, int out_size, void* d_ws, size_t ws_size,
                              hipStream_t stream) {
    const float* z       = (const float*)d_in[0];   // [8192,128]
    const int*   y_idx   = (const int*)  d_in[1];   // [8192]
    const float* proxies = (const float*)d_in[2];   // [93,128]
    const int*   y_map   = (const int*)  d_in[3];   // [62]
    float* out = (float*)d_out;

    // Workspace carve-up (~2.4 MB total, 16B aligned slabs; dominated by the
    // f16 mirror of z -- trivially L2-resident on MI455X's 192 MB L2).
    char* w = (char*)d_ws;
    size_t off = 0;
    auto take = [&](size_t bytes) { char* p = w + off; off = (off + bytes + 15) & ~(size_t)15; return p; };
    _Float16* zh   = (_Float16*)take((size_t)B_N * Z_D * 2);
    _Float16* prxh = (_Float16*)take((size_t)PPAD * Z_D * 2);
    float* zz      = (float*)take(B_N * 4);
    float* zsum    = (float*)take(B_N * 4);
    int*   yrel    = (int*)  take(B_N * 4);
    float* prx     = (float*)take((size_t)PPAD * Z_D * 4);
    float* psum    = (float*)take(PPAD * 4);
    float* pnrm2   = (float*)take(PPAD * 4);
    unsigned long long* pmin = (unsigned long long*)take(APAD * 8);
    unsigned long long* pmax = (unsigned long long*)take(APAD * 8);
    int*   pidx    = (int*)  take(APAD * 4);
    float* a_sum   = (float*)take(APAD * 4);
    float* a_nrm2  = (float*)take(APAD * 4);
    int*   a_cls   = (int*)  take(APAD * 4);
    float* loss    = (float*)take(APAD * 4);

    k_init<<<(APAD + 255) / 256, 256, 0, stream>>>(pmin, pmax, loss);
    k_prep_z<<<B_N, 128, 0, stream>>>(z, y_idx, y_map, zz, zsum, yrel, zh);
    k_prep_prx<<<PPAD, 128, 0, stream>>>(proxies, prx, prxh, psum, pnrm2);
    k_nearest_proxy<<<MT, 32, 0, stream>>>(zh, prxh, zz, zsum, psum, pnrm2, pmin);
    k_gather_anchor<<<(APAD + 255) / 256, 256, 0, stream>>>(pmin, psum, pnrm2, yrel,
                                                            pidx, a_sum, a_nrm2, a_cls);
    k_hard_pos<<<dim3(MT, B_N / 64), 32, 0, stream>>>(zh, prxh, zz, zsum, yrel, pidx,
                                                      a_sum, a_nrm2, a_cls, pmax);
    k_loss<<<NA, 128, 0, stream>>>(z, prx, zz, zsum, psum, pnrm2, pmax, loss);
    k_reduce<<<1, 256, 0, stream>>>(loss, out);
}
